// EmpathyAttention_13460427506063
// MI455X (gfx1250) — matrix operand order repaired and checked by reading, not compiled, not run on hardware
//
#include <hip/hip_runtime.h>
#include <hip/hip_bf16.h>

// ------------------------------------------------------------------
// EmpathyAttention for MI455X (gfx1250): all GEMMs via
// v_wmma_f32_16x16x32_bf16, flash-style online-softmax attention with
// block-cooperative LDS staging of K/V tiles.
// The cosine "empathy" bias is constant per (b,h) over the softmax
// axis -> softmax-invariant -> elided.
// Fragment loads are hoisted ahead of each WMMA burst so the LDS/VMEM
// loads overlap instead of serializing on a shared register group.
// ------------------------------------------------------------------

typedef __attribute__((ext_vector_type(16))) __bf16 bf16x16;
typedef __attribute__((ext_vector_type(8)))  __bf16 bf16x8;
typedef __attribute__((ext_vector_type(4)))  __bf16 bf16x4;
typedef __attribute__((ext_vector_type(8)))  float  f32x8;

union Frag { bf16x16 v; bf16x8 h[2]; };

#define DMODEL 1024
#define NHEADS 16
#define DKH    64
#define NBATCH 4
#define SEQ    1024
#define MTOK   (NBATCH * SEQ)   // 4096

static __device__ __forceinline__
f32x8 wmma_bf16(const Frag& a, const Frag& b, f32x8 c) {
  return __builtin_amdgcn_wmma_f32_16x16x32_bf16(
      /*neg_a=*/false, a.v, /*neg_b=*/false, b.v,
      /*c_mod=*/(short)0, c, /*reuse_a=*/false, /*reuse_b=*/false);
}

// ---------------- fp32 -> bf16 conversion (vectorized x4) ----------------
__global__ void f32_to_bf16_kernel(const float* __restrict__ in,
                                   __bf16* __restrict__ out) {
  int i = blockIdx.x * blockDim.x + threadIdx.x;
  float4 f = ((const float4*)in)[i];
  bf16x4 o;
  o[0] = (__bf16)f.x; o[1] = (__bf16)f.y; o[2] = (__bf16)f.z; o[3] = (__bf16)f.w;
  ((bf16x4*)out)[i] = o;
}

// ------------- weight transpose + convert: Wt[n,k] = (bf16)W[k,n] ---------
__global__ void w_transpose_bf16_kernel(const float* __restrict__ W,
                                        __bf16* __restrict__ Wt) {
  int idx = blockIdx.x * blockDim.x + threadIdx.x;     // 0 .. 1M-1
  int k = idx >> 10;
  int n = idx & (DMODEL - 1);
  Wt[(size_t)n * DMODEL + k] = (__bf16)W[idx];
}

// ------------------------------------------------------------------
// Generic bf16 WMMA GEMM: C[M,N] = A[M,K] * Bt[N,K]^T + bias[N]
// Per wave: 64(M) x 64(N) tile (16 accumulators); block = 4 waves -> 64x256.
// All 8 fragment loads issued before the 16-WMMA burst each k-step.
// MODE 0: store bf16 head-split  [b,h,s,dk]   (Q, K)
// MODE 1: store bf16 transposed  [b,h,dk,s]   (V)  -- contiguous b128 stores
// MODE 2: store fp32 row-major   [M, DMODEL]  (final output)
// ------------------------------------------------------------------
template <int MODE>
__global__ __launch_bounds__(128)
void gemm_bf16_wmma_kernel(const __bf16* __restrict__ A,
                           const __bf16* __restrict__ Bt,
                           const float*  __restrict__ bias,
                           void* __restrict__ Cout, int Kdim) {
  const int lane = threadIdx.x & 31;
  const int wave = threadIdx.x >> 5;
  const int half = lane >> 4;        // 0/1 : which 16-lane half
  const int ln   = lane & 15;
  const int m0 = blockIdx.y * 64;
  const int n0 = blockIdx.x * 256 + wave * 64;

  const f32x8 zero8 = {0.f,0.f,0.f,0.f,0.f,0.f,0.f,0.f};
  f32x8 acc[4][4];
#pragma unroll
  for (int mi = 0; mi < 4; ++mi)
#pragma unroll
    for (int ci = 0; ci < 4; ++ci) acc[mi][ci] = zero8;

  for (int kk = 0; kk < Kdim; kk += 32) {
    Frag a[4], b[4];
#pragma unroll
    for (int mi = 0; mi < 4; ++mi) {
      const __bf16* ap = A + (size_t)(m0 + mi * 16 + ln) * Kdim + kk;
      a[mi].h[0] = *(const bf16x8*)(ap + 16 * half);       // K = 16h .. 16h+7
      a[mi].h[1] = *(const bf16x8*)(ap + 16 * half + 8);   // K = 16h+8 ..
    }
#pragma unroll
    for (int ci = 0; ci < 4; ++ci) {
      const __bf16* bp = Bt + (size_t)(n0 + ci * 16 + ln) * Kdim + kk + 16 * half;
      b[ci].h[0] = *(const bf16x8*)(bp);                   // K = 16h .. 16h+7
      b[ci].h[1] = *(const bf16x8*)(bp + 8);               // K = 16h+8 ..
    }
#pragma unroll
    for (int ci = 0; ci < 4; ++ci)
#pragma unroll
      for (int mi = 0; mi < 4; ++mi)
        acc[mi][ci] = wmma_bf16(a[mi], b[ci], acc[mi][ci]);
  }

#pragma unroll
  for (int ci = 0; ci < 4; ++ci) {
    const int n  = n0 + ci * 16 + ln;
    const float bv = bias ? bias[n] : 0.0f;
#pragma unroll
    for (int mi = 0; mi < 4; ++mi) {
      if (MODE == 1) {
        // Vt[b,h,d,s]: lane's 8 acc rows are contiguous s -> one b128 store
        const int mbase = m0 + mi * 16 + 8 * half;
        const int bidx  = mbase >> 10;
        const int sbase = mbase & (SEQ - 1);
        const int hh = n >> 6, d = n & (DKH - 1);
        bf16x8 pk;
#pragma unroll
        for (int r = 0; r < 8; ++r) pk[r] = (__bf16)(acc[mi][ci][r] + bv);
        *(bf16x8*)((__bf16*)Cout +
                   ((size_t)((bidx * NHEADS + hh) * DKH + d)) * SEQ + sbase) = pk;
      } else {
#pragma unroll
        for (int r = 0; r < 8; ++r) {
          const int m = m0 + mi * 16 + r + 8 * half;
          const float v = acc[mi][ci][r] + bv;
          if (MODE == 0) {
            const int bidx = m >> 10, s = m & (SEQ - 1);
            const int hh = n >> 6,   d = n & (DKH - 1);
            ((__bf16*)Cout)[((size_t)((bidx * NHEADS + hh) * SEQ + s)) * DKH + d]
                = (__bf16)v;
          } else {
            ((float*)Cout)[(size_t)m * DMODEL + n] = v;
          }
        }
      }
    }
  }
}

// ------------------------------------------------------------------
// Flash attention: block = 4 waves sharing one (b,h), 64 queries total.
// K/V tiles (32 keys) staged cooperatively in LDS each step (4KB+4KB);
// scores via WMMA, online softmax in acc layout, P through per-wave LDS
// (acc layout -> A-fragment layout), P@V via WMMA.
// ------------------------------------------------------------------
__global__ __launch_bounds__(128)
void attn_flash_kernel(const __bf16* __restrict__ Q,
                       const __bf16* __restrict__ K,
                       const __bf16* __restrict__ Vt,
                       const int*    __restrict__ mask,
                       __bf16* __restrict__ attn) {
  __shared__ __align__(16) __bf16 ldsK[32 * DKH];   // [key 0..31][dk 0..63]
  __shared__ __align__(16) __bf16 ldsV[DKH * 32];   // [d 0..63][key 0..31]
  __shared__ __align__(16) __bf16 plds[4][16 * 32]; // per-wave P staging

  const int tid  = threadIdx.x;
  const int lane = tid & 31;
  const int wave = tid >> 5;
  const int half = lane >> 4;
  const int ln   = lane & 15;
  // block -> (b, h, 64-query group); wave -> 16-query tile within group
  const int qg = blockIdx.x & 15;
  const int h  = (blockIdx.x >> 4) & (NHEADS - 1);
  const int b  = blockIdx.x >> 8;
  const int q0 = qg * 64 + wave * 16;

  const __bf16* Qh  = Q  + ((size_t)(b * NHEADS + h) * SEQ + q0) * DKH;
  const __bf16* Kh  = K  +  (size_t)(b * NHEADS + h) * SEQ * DKH;
  const __bf16* Vth = Vt +  (size_t)(b * NHEADS + h) * DKH * SEQ;
  const int*    mb  = mask + (size_t)b * SEQ * SEQ;
  __bf16* myp = plds[wave];

  // Q A-fragments (dk split 0..31 / 32..63), resident for the whole loop
  Frag aq0, aq1;
  {
    const __bf16* qp = Qh + ln * DKH;
    aq0.h[0] = *(const bf16x8*)(qp + 16 * half);
    aq0.h[1] = *(const bf16x8*)(qp + 16 * half + 8);
    aq1.h[0] = *(const bf16x8*)(qp + 32 + 16 * half);
    aq1.h[1] = *(const bf16x8*)(qp + 32 + 16 * half + 8);
  }

  const f32x8 zero8 = {0.f,0.f,0.f,0.f,0.f,0.f,0.f,0.f};
  float mx[8], ls[8];
#pragma unroll
  for (int r = 0; r < 8; ++r) { mx[r] = -3.0e38f; ls[r] = 0.0f; }
  f32x8 o[4];
#pragma unroll
  for (int t = 0; t < 4; ++t) o[t] = zero8;

  for (int kt = 0; kt < SEQ; kt += 32) {
    // ---- cooperative staging of K (32x64) and V (64x32) tiles ----
    __syncthreads();   // previous-step fragment reads complete
    {
      const int krow = tid >> 2, kcol = (tid & 3) * 16;        // 32 x 4
      const __bf16* ksrc = Kh + (size_t)(kt + krow) * DKH + kcol;
      *(bf16x8*)&ldsK[krow * DKH + kcol]     = *(const bf16x8*)(ksrc);
      *(bf16x8*)&ldsK[krow * DKH + kcol + 8] = *(const bf16x8*)(ksrc + 8);
      const int vrow = tid >> 1, vcol = (tid & 1) * 16;        // 64 x 2
      const __bf16* vsrc = Vth + (size_t)vrow * SEQ + kt + vcol;
      *(bf16x8*)&ldsV[vrow * 32 + vcol]     = *(const bf16x8*)(vsrc);
      *(bf16x8*)&ldsV[vrow * 32 + vcol + 8] = *(const bf16x8*)(vsrc + 8);
    }
    if (kt + 32 < SEQ) {
      __builtin_prefetch(Kh  + (size_t)(kt + 32) * DKH, 0, 1);
      __builtin_prefetch(Vth + (size_t)(kt + 32), 0, 1);
    }
    __syncthreads();   // tiles visible to all waves

    // ---- scores = Q * K^T: load all 4 K-frags, then 4-WMMA burst ----
    Frag bk[2][2];
#pragma unroll
    for (int st = 0; st < 2; ++st) {
      const __bf16* kp = &ldsK[(st * 16 + ln) * DKH];
      bk[st][0].h[0] = *(const bf16x8*)(kp + 16 * half);
      bk[st][0].h[1] = *(const bf16x8*)(kp + 16 * half + 8);
      bk[st][1].h[0] = *(const bf16x8*)(kp + 32 + 16 * half);
      bk[st][1].h[1] = *(const bf16x8*)(kp + 32 + 16 * half + 8);
    }
    f32x8 sc[2];
#pragma unroll
    for (int st = 0; st < 2; ++st) {
      sc[st] = wmma_bf16(aq0, bk[st][0], zero8);
      sc[st] = wmma_bf16(aq1, bk[st][1], sc[st]);
    }

    // ---- scale + mask ----
#pragma unroll
    for (int st = 0; st < 2; ++st) {
      const int key = kt + st * 16 + ln;
#pragma unroll
      for (int r = 0; r < 8; ++r) {
        const int q = q0 + r + 8 * half;
        float v = sc[st][r] * 0.125f;                 // 1/sqrt(64)
        if (mb[(size_t)q * SEQ + key] == 0) v = -1.0e9f;
        sc[st][r] = v;
      }
    }

    // ---- online softmax (row stats live per acc-VGPR) ----
    float corr[8];
#pragma unroll
    for (int r = 0; r < 8; ++r) {
      float rm = fmaxf(sc[0][r], sc[1][r]);
#pragma unroll
      for (int off = 8; off >= 1; off >>= 1)
        rm = fmaxf(rm, __shfl_xor(rm, off, 32));
      const float mn = fmaxf(mx[r], rm);
      corr[r] = __expf(mx[r] - mn);
      mx[r] = mn;
      sc[0][r] = __expf(sc[0][r] - mn);
      sc[1][r] = __expf(sc[1][r] - mn);
      float rs = sc[0][r] + sc[1][r];
#pragma unroll
      for (int off = 8; off >= 1; off >>= 1)
        rs += __shfl_xor(rs, off, 32);
      ls[r] = ls[r] * corr[r] + rs;
    }
#pragma unroll
    for (int t = 0; t < 4; ++t)
#pragma unroll
      for (int r = 0; r < 8; ++r) o[t][r] *= corr[r];

    // ---- stage P (acc layout) -> LDS -> A-fragment layout ----
#pragma unroll
    for (int st = 0; st < 2; ++st)
#pragma unroll
      for (int r = 0; r < 8; ++r)
        myp[(r + 8 * half) * 32 + st * 16 + ln] = (__bf16)sc[st][r];
    asm volatile("s_wait_dscnt 0" ::: "memory");   // same-wave DS ordering
    Frag pf;
    pf.h[0] = *(const bf16x8*)(myp + ln * 32 + 16 * half);
    pf.h[1] = *(const bf16x8*)(myp + ln * 32 + 16 * half + 8);

    // ---- O += P * V: load all 4 V-frags, then 4-WMMA burst ----
    Frag bvv[4];
#pragma unroll
    for (int t = 0; t < 4; ++t) {
      const __bf16* vp = &ldsV[(t * 16 + ln) * 32 + 16 * half];
      bvv[t].h[0] = *(const bf16x8*)(vp);
      bvv[t].h[1] = *(const bf16x8*)(vp + 8);
    }
#pragma unroll
    for (int t = 0; t < 4; ++t)
      o[t] = wmma_bf16(pf, bvv[t], o[t]);
  }

  // ---- epilogue: normalize + store merged-head bf16 [B,S,DMODEL] ----
#pragma unroll
  for (int t = 0; t < 4; ++t) {
    const int d = t * 16 + ln;
#pragma unroll
    for (int r = 0; r < 8; ++r) {
      const int q = q0 + r + 8 * half;
      const float v = o[t][r] / ls[r];
      attn[((size_t)(b * SEQ + q)) * DMODEL + h * DKH + d] = (__bf16)v;
    }
  }
}

// ------------------------------------------------------------------
extern "C" void kernel_launch(void* const* d_in, const int* in_sizes, int n_in,
                              void* d_out, int out_size, void* d_ws, size_t ws_size,
                              hipStream_t stream) {
  (void)in_sizes; (void)n_in; (void)out_size; (void)ws_size;
  const float* q   = (const float*)d_in[0];
  const float* k   = (const float*)d_in[1];
  const float* v   = (const float*)d_in[2];
  const int*   msk = (const int*)  d_in[3];
  const float* Wq  = (const float*)d_in[4];
  const float* bq  = (const float*)d_in[5];
  const float* Wk  = (const float*)d_in[6];
  const float* bk  = (const float*)d_in[7];
  const float* Wv  = (const float*)d_in[8];
  const float* bv  = (const float*)d_in[9];
  const float* Wo  = (const float*)d_in[10];
  const float* bo  = (const float*)d_in[11];

  char* ws = (char*)d_ws;
  const size_t MB = 1024u * 1024u;
  __bf16* qb   = (__bf16*)(ws + 0 * MB);    // 8 MB each: bf16 [4096,1024]
  __bf16* kb   = (__bf16*)(ws + 8 * MB);
  __bf16* vb   = (__bf16*)(ws + 16 * MB);
  __bf16* WqT  = (__bf16*)(ws + 24 * MB);   // 2 MB each: bf16 [N,K]
  __bf16* WkT  = (__bf16*)(ws + 26 * MB);
  __bf16* WvT  = (__bf16*)(ws + 28 * MB);
  __bf16* WoT  = (__bf16*)(ws + 30 * MB);
  __bf16* Qh   = (__bf16*)(ws + 32 * MB);   // 8 MB: bf16 [B,H,S,dk]
  __bf16* Kh   = (__bf16*)(ws + 40 * MB);   // 8 MB: bf16 [B,H,S,dk]
  __bf16* Vth  = (__bf16*)(ws + 48 * MB);   // 8 MB: bf16 [B,H,dk,S]
  __bf16* attn = (__bf16*)(ws + 56 * MB);   // 8 MB: bf16 [4096,1024]

  // 1) fp32 -> bf16 conversions (x4 vectorized): 4M elems each
  f32_to_bf16_kernel<<<4096, 256, 0, stream>>>(q, qb);
  f32_to_bf16_kernel<<<4096, 256, 0, stream>>>(k, kb);
  f32_to_bf16_kernel<<<4096, 256, 0, stream>>>(v, vb);

  // 2) weight transpose+convert: 1M elems each
  w_transpose_bf16_kernel<<<4096, 256, 0, stream>>>(Wq, WqT);
  w_transpose_bf16_kernel<<<4096, 256, 0, stream>>>(Wk, WkT);
  w_transpose_bf16_kernel<<<4096, 256, 0, stream>>>(Wv, WvT);
  w_transpose_bf16_kernel<<<4096, 256, 0, stream>>>(Wo, WoT);

  // 3) projections (WMMA GEMMs). Grid: N/256 x M/64
  dim3 ggrid(DMODEL / 256, MTOK / 64);   // (4, 64)
  gemm_bf16_wmma_kernel<0><<<ggrid, 128, 0, stream>>>(qb, WqT, bq, Qh,  DMODEL);
  gemm_bf16_wmma_kernel<0><<<ggrid, 128, 0, stream>>>(kb, WkT, bk, Kh,  DMODEL);
  gemm_bf16_wmma_kernel<1><<<ggrid, 128, 0, stream>>>(vb, WvT, bv, Vth, DMODEL);

  // 4) flash attention: 1024 blocks x 4 waves (one (b,h) per block group)
  attn_flash_kernel<<<1024, 128, 0, stream>>>(Qh, Kh, Vth, msk, attn);

  // 5) output projection -> fp32 d_out
  gemm_bf16_wmma_kernel<2><<<ggrid, 128, 0, stream>>>(attn, WoT, bo, d_out, DMODEL);
}
